// Block_9414568312920
// MI455X (gfx1250) — matrix operand order repaired and checked by reading, not compiled, not run on hardware
//
#include <hip/hip_runtime.h>
#include <hip/hip_bf16.h>
#include <math.h>

// ---------------- problem constants ----------------
#define BB   32
#define N0   513
#define TT   10
#define NN   523            // N0 + T
#define CC   768
#define HH   12
#define DH   64
#define HID  3072
#define MM   (BB * NN)      // 16736 = 523 * 32
#define SCALE 0.125f        // Dh^-0.5
#define TILE_K 64

typedef __attribute__((ext_vector_type(16))) __bf16 v16bf;
typedef __attribute__((ext_vector_type(8)))  float  v8f;

union BFrag {
    v16bf v;
    unsigned short h[16];
    unsigned int   u[8];
};

__device__ __forceinline__ int imin(int a, int b) { return a < b ? a : b; }

__device__ __forceinline__ unsigned short f2bf(float f) {
    union { float f; unsigned u; } x; x.f = f;
    unsigned r = x.u + 0x7FFFu + ((x.u >> 16) & 1u);   // round-nearest-even
    return (unsigned short)(r >> 16);
}

__device__ __forceinline__ v8f wmma_bf16(const BFrag& a, const BFrag& b, v8f c) {
    return __builtin_amdgcn_wmma_f32_16x16x32_bf16(false, a.v, false, b.v,
                                                   (short)0, c, false, false);
}

__device__ __forceinline__ float redmax16(float x) {
    x = fmaxf(x, __shfl_xor(x, 1, 32));
    x = fmaxf(x, __shfl_xor(x, 2, 32));
    x = fmaxf(x, __shfl_xor(x, 4, 32));
    x = fmaxf(x, __shfl_xor(x, 8, 32));
    return x;
}
__device__ __forceinline__ float redsum16(float x) {
    x += __shfl_xor(x, 1, 32);
    x += __shfl_xor(x, 2, 32);
    x += __shfl_xor(x, 4, 32);
    x += __shfl_xor(x, 8, 32);
    return x;
}

// ---------------- elementwise kernels ----------------

__global__ void cvt_bf16_kernel(const float* __restrict__ in,
                                unsigned short* __restrict__ out, int n) {
    int i = blockIdx.x * 256 + threadIdx.x;
    if (i < n) out[i] = f2bf(in[i]);
}

// build fused token stream x' = [x[:, :1], prompt_tokens, x[:, 1:]]  -> fp32 [B*N, C]
__global__ void fuse_kernel(const float* __restrict__ x,
                            const float* __restrict__ gf,
                            const float* __restrict__ tp,
                            const float* __restrict__ pe,
                            const float* __restrict__ scaler,
                            float* __restrict__ xf) {
    long long i = (long long)blockIdx.x * 256 + threadIdx.x;
    const long long total = (long long)MM * CC;
    if (i >= total) return;
    int c = (int)(i % CC);
    long long rn = i / CC;
    int n = (int)(rn % NN);
    int b = (int)(rn / NN);
    float v;
    if (n == 0) {
        v = x[((long long)b * N0) * CC + c];
    } else if (n <= TT) {
        int t = n - 1;
        v = pe[t * CC + c] + scaler[0] * gf[(long long)b * CC + c] + tp[t * CC + c];
    } else {
        v = x[((long long)b * N0 + (n - TT)) * CC + c];
    }
    xf[i] = v;
}

// LayerNorm over C=768, output bf16. One 256-thread block per row.
__global__ void ln_kernel(const float* __restrict__ xf,
                          const float* __restrict__ g,
                          const float* __restrict__ bta,
                          unsigned short* __restrict__ out) {
    const int row = blockIdx.x;
    const int tid = threadIdx.x;
    const float* xr = xf + (long long)row * CC;
    float v[3];
    float s = 0.f;
#pragma unroll
    for (int k = 0; k < 3; ++k) { v[k] = xr[tid + k * 256]; s += v[k]; }
    __shared__ float red[256];
    red[tid] = s; __syncthreads();
    for (int off = 128; off > 0; off >>= 1) {
        if (tid < off) red[tid] += red[tid + off];
        __syncthreads();
    }
    float mean = red[0] * (1.0f / CC);
    __syncthreads();
    float sq = 0.f;
#pragma unroll
    for (int k = 0; k < 3; ++k) { float d = v[k] - mean; sq += d * d; }
    red[tid] = sq; __syncthreads();
    for (int off = 128; off > 0; off >>= 1) {
        if (tid < off) red[tid] += red[tid + off];
        __syncthreads();
    }
    float rstd = rsqrtf(red[0] * (1.0f / CC) + 1e-5f);
#pragma unroll
    for (int k = 0; k < 3; ++k) {
        int c = tid + k * 256;
        out[(long long)row * CC + c] = f2bf((v[k] - mean) * rstd * g[c] + bta[c]);
    }
}

// ---------------- WMMA GEMM with async-LDS staged A ----------------
// out[m, n] = act[m, :] . W[n, :]   (act: [M,K] bf16 row-major, W: [Ncols,K] bf16 row-major)
// Block: 256 thr = 8 waves; patch 32 rows x 256 cols; each wave does 16x64 (4 accums).
// A tile (32 x 64 bf16 = 4KB) staged in LDS via global_load_async_to_lds_b128, double-buffered.
// mode 0: fp32 out = acc + bias[n] + res[m,n]
// mode 1: bf16 out = acc (+ bias if given)
// mode 2: bf16 out = gelu(acc + bias[n])   (exact gelu)
__global__ void gemm_bf16_kernel(const unsigned short* __restrict__ A, int lda,
                                 const unsigned short* __restrict__ W, int ldw,
                                 const float* __restrict__ bias,
                                 const float* __restrict__ res,
                                 float* __restrict__ outF,
                                 unsigned short* __restrict__ outH,
                                 int ldo, int K, int mode) {
    __shared__ __align__(16) unsigned short tileA[2][32 * TILE_K];   // 2 x 4KB

    const int tid  = threadIdx.x;
    const int lane = tid & 31;
    const int wv   = tid >> 5;
    const int hl = lane >> 4, nl = lane & 15;
    const int rowTile = blockIdx.x * 32;
    const int row0 = rowTile + (wv >> 2) * 16;           // wave's 16 output rows
    const int col0 = blockIdx.y * 256 + (wv & 3) * 64;   // wave's 64 output cols

    // staging: thread t loads 16B: row = t/8, halfs [(t%8)*8 .. +7]
    const int sRow = tid >> 3;
    const int sCol = (tid & 7) * 8;
    const unsigned short* gA = A + (long long)(rowTile + sRow) * lda + sCol;
    const unsigned ldsDst0 =
        (unsigned)(unsigned long long)(const void*)(&tileA[0][sRow * TILE_K + sCol]);
    const unsigned ldsDst1 =
        (unsigned)(unsigned long long)(const void*)(&tileA[1][sRow * TILE_K + sCol]);

    const unsigned short* Wr = W + (long long)(col0 + nl) * ldw;     // B-frag col n = lane%16

    v8f acc[4] = {{}, {}, {}, {}};

    // prologue: stage first tile into buffer 0
    {
        unsigned long long ga = (unsigned long long)(const void*)gA;
        asm volatile("global_load_async_to_lds_b128 %0, %1, off"
                     :: "v"(ldsDst0), "v"(ga) : "memory");
    }

    const int nStages = K / TILE_K;
    for (int st = 0; st < nStages; ++st) {
        const int kb = st * TILE_K;
        const int buf = st & 1;
        // wait for my async writes into current buffer, then make visible to all waves
        asm volatile("s_wait_asynccnt 0" ::: "memory");
        __syncthreads();
        // prefetch next stage into other buffer (overlaps with WMMA below)
        if (st + 1 < nStages) {
            unsigned long long ga = (unsigned long long)(const void*)(gA + kb + TILE_K);
            asm volatile("global_load_async_to_lds_b128 %0, %1, off"
                         :: "v"(buf ? ldsDst0 : ldsDst1), "v"(ga) : "memory");
        }

        const unsigned short* tA = &tileA[buf][((wv >> 2) * 16 + nl) * TILE_K];
#pragma unroll
        for (int s = 0; s < 2; ++s) {                    // two 32-wide k-steps
            BFrag a;
            const uint4* ap = (const uint4*)(tA + s * 32);
            uint4 lo = ap[2 * hl];      // halfs s*32 + hl*16 .. +7  (k = hl*16 + 2v)
            uint4 hi = ap[2 * hl + 1];  // halfs s*32 + hl*16 + 8    (k = hl*16 + 8 + ...)
            a.u[0] = lo.x; a.u[1] = lo.y; a.u[2] = lo.z; a.u[3] = lo.w;
            a.u[4] = hi.x; a.u[5] = hi.y; a.u[6] = hi.z; a.u[7] = hi.w;
#pragma unroll
            for (int cn = 0; cn < 4; ++cn) {
                BFrag b;
                const uint4* wp = (const uint4*)(Wr + (long long)cn * 16 * ldw
                                                 + kb + s * 32 + hl * 16);
                uint4 w0 = wp[0], w1 = wp[1];            // k = hl*16 + 2j
                b.u[0] = w0.x; b.u[1] = w0.y; b.u[2] = w0.z; b.u[3] = w0.w;
                b.u[4] = w1.x; b.u[5] = w1.y; b.u[6] = w1.z; b.u[7] = w1.w;
                acc[cn] = wmma_bf16(a, b, acc[cn]);
            }
        }
        __syncthreads();   // all waves done reading buf before it is refilled next+1 stage
    }

#pragma unroll
    for (int cn = 0; cn < 4; ++cn) {
        const int col = col0 + cn * 16 + nl;
        const float bs = bias ? bias[col] : 0.0f;
#pragma unroll
        for (int r = 0; r < 8; ++r) {
            int m = r + 8 * hl;                          // C/D layout
            long long o = (long long)(row0 + m) * ldo + col;
            float val = acc[cn][r] + bs;
            if (mode == 0) {
                outF[o] = val + res[o];
            } else if (mode == 1) {
                outH[o] = f2bf(val);
            } else {
                float gl = 0.5f * val * (1.0f + erff(val * 0.70710678118654752f));
                outH[o] = f2bf(gl);
            }
        }
    }
}

// ---------------- flash attention ----------------
// qkv: bf16 [B*N, 3C]; out: bf16 [B*N, C]. One wave per (qtile, head, batch).
__global__ void attn_kernel(const unsigned short* __restrict__ qkv,
                            unsigned short* __restrict__ out) {
    const int l  = threadIdx.x;
    const int hl = l >> 4, nl = l & 15;
    const int qt = blockIdx.x;      // 0..32 (33 q tiles)
    const int h  = blockIdx.y;      // 0..11
    const int b  = blockIdx.z;      // 0..31
    const int rb = b * NN;
    const int q0 = qt * 16;
    const int cq = h * DH;
    const int ck = CC + h * DH;
    const int cv = 2 * CC + h * DH;

    __shared__ __align__(4) unsigned short plds[16 * 32];

    BFrag qa0, qa1;
    {
        int row = rb + imin(q0 + nl, NN - 1);
        const unsigned short* qr = qkv + (long long)row * (3 * CC) + cq;
#pragma unroll
        for (int v = 0; v < 8; ++v) {
            int k = hl * 16 + (v << 1);
            qa0.u[v] = *(const unsigned int*)(qr + k);
            qa1.u[v] = *(const unsigned int*)(qr + 32 + k);
        }
    }

    v8f acc[4] = {{}, {}, {}, {}};
    float rmax[8], rsum[8];
#pragma unroll
    for (int r = 0; r < 8; ++r) { rmax[r] = -1e30f; rsum[r] = 0.f; }

    for (int kc = 0; kc < 17; ++kc) {           // 17 chunks x 32 keys = 544 >= 523
        const int kbase = kc * 32;
        v8f s[2];
#pragma unroll
        for (int t = 0; t < 2; ++t) {
            int key = kbase + t * 16 + nl;      // B-frag column n = lane%16
            const unsigned short* kr =
                qkv + (long long)(rb + imin(key, NN - 1)) * (3 * CC) + ck;
            BFrag b0, b1;
#pragma unroll
            for (int j = 0; j < 8; ++j) {
                int kk = hl * 16 + (j << 1);
                b0.u[j] = *(const unsigned int*)(kr + kk);
                b1.u[j] = *(const unsigned int*)(kr + 32 + kk);
            }
            v8f st = {};
            st = wmma_bf16(qa0, b0, st);
            st = wmma_bf16(qa1, b1, st);
            bool valid = (key < NN);
#pragma unroll
            for (int r = 0; r < 8; ++r)
                s[t][r] = valid ? st[r] * SCALE : -1e30f;
        }

        // online softmax update + stage P into LDS
#pragma unroll
        for (int r = 0; r < 8; ++r) {
            float tm = redmax16(fmaxf(s[0][r], s[1][r]));
            float nm = fmaxf(rmax[r], tm);
            float al = __expf(rmax[r] - nm);
            rmax[r] = nm;
            float p0 = __expf(s[0][r] - nm);
            float p1 = __expf(s[1][r] - nm);
            rsum[r] = rsum[r] * al + redsum16(p0 + p1);
            acc[0][r] *= al; acc[1][r] *= al; acc[2][r] *= al; acc[3][r] *= al;
            int m = r + 8 * hl;
            plds[m * 32 + nl]      = f2bf(p0);
            plds[m * 32 + 16 + nl] = f2bf(p1);
        }
        __syncthreads();

        BFrag pa;                                  // P as A-fragment
#pragma unroll
        for (int v = 0; v < 8; ++v) {
            int kk = hl * 16 + (v << 1);
            pa.u[v] = *(const unsigned int*)(plds + nl * 32 + kk);
        }
        __syncthreads();

#pragma unroll
        for (int vt = 0; vt < 4; ++vt) {           // P @ V over Dh=64
            BFrag vb;
#pragma unroll
            for (int j = 0; j < 8; ++j) {
                int kk = hl * 16 + (j << 1);
                int k0 = imin(kbase + kk,     NN - 1);
                int k1 = imin(kbase + kk + 1, NN - 1);
                vb.h[2 * j]     = qkv[(long long)(rb + k0) * (3 * CC) + cv + vt * 16 + nl];
                vb.h[2 * j + 1] = qkv[(long long)(rb + k1) * (3 * CC) + cv + vt * 16 + nl];
            }
            acc[vt] = wmma_bf16(pa, vb, acc[vt]);
        }
    }

#pragma unroll
    for (int r = 0; r < 8; ++r) {
        int m = r + 8 * hl;
        int qg = q0 + m;
        if (qg < NN) {
            float inv = 1.0f / rsum[r];
            long long o = (long long)(rb + qg) * CC + h * DH;
            out[o + 0 * 16 + nl] = f2bf(acc[0][r] * inv);
            out[o + 1 * 16 + nl] = f2bf(acc[1][r] * inv);
            out[o + 2 * 16 + nl] = f2bf(acc[2][r] * inv);
            out[o + 3 * 16 + nl] = f2bf(acc[3][r] * inv);
        }
    }
}

// ---------------- host-side orchestration ----------------
extern "C" void kernel_launch(void* const* d_in, const int* in_sizes, int n_in,
                              void* d_out, int out_size, void* d_ws, size_t ws_size,
                              hipStream_t stream) {
    const float* x      = (const float*)d_in[0];
    const float* gf     = (const float*)d_in[1];
    const float* tp     = (const float*)d_in[2];
    const float* pe     = (const float*)d_in[3];
    const float* sc1    = (const float*)d_in[4];
    const float* g1     = (const float*)d_in[5];
    const float* b1     = (const float*)d_in[6];
    const float* w_qkv  = (const float*)d_in[7];
    const float* w_proj = (const float*)d_in[8];
    const float* b_proj = (const float*)d_in[9];
    const float* g2     = (const float*)d_in[10];
    const float* b2     = (const float*)d_in[11];
    const float* w_fc1  = (const float*)d_in[12];
    const float* b_fc1  = (const float*)d_in[13];
    const float* w_fc2  = (const float*)d_in[14];
    const float* b_fc2  = (const float*)d_in[15];

    size_t off = 0;
    char* ws = (char*)d_ws;
    auto carve = [&](size_t bytes) -> char* {
        off = (off + 255) & ~(size_t)255;
        char* p = ws + off;
        off += bytes;
        return p;
    };
    float*          xf   = (float*)carve((size_t)MM * CC * 4);          // residual stream
    unsigned short* abf  = (unsigned short*)carve((size_t)MM * CC * 2); // ln-out / attn-out
    unsigned short* big  = (unsigned short*)carve((size_t)MM * HID * 2);// qkv then mlp-hidden
    unsigned short* wqb  = (unsigned short*)carve((size_t)3 * CC * CC * 2);
    unsigned short* wpb  = (unsigned short*)carve((size_t)CC * CC * 2);
    unsigned short* w1b  = (unsigned short*)carve((size_t)HID * CC * 2);
    unsigned short* w2b  = (unsigned short*)carve((size_t)CC * HID * 2);
    if (off > ws_size) return;

    {
        int n;
        n = 3 * CC * CC; cvt_bf16_kernel<<<(n + 255) / 256, 256, 0, stream>>>(w_qkv, wqb, n);
        n = CC * CC;     cvt_bf16_kernel<<<(n + 255) / 256, 256, 0, stream>>>(w_proj, wpb, n);
        n = HID * CC;    cvt_bf16_kernel<<<(n + 255) / 256, 256, 0, stream>>>(w_fc1, w1b, n);
        n = CC * HID;    cvt_bf16_kernel<<<(n + 255) / 256, 256, 0, stream>>>(w_fc2, w2b, n);
    }

    {
        long long total = (long long)MM * CC;
        fuse_kernel<<<(unsigned)((total + 255) / 256), 256, 0, stream>>>(x, gf, tp, pe, sc1, xf);
    }

    ln_kernel<<<MM, 256, 0, stream>>>(xf, g1, b1, abf);

    // QKV: [M,768] x [2304,768]^T -> bf16 [M,2304]
    gemm_bf16_kernel<<<dim3(MM / 32, (3 * CC) / 256), 256, 0, stream>>>(
        abf, CC, wqb, CC, nullptr, nullptr, nullptr, big, 3 * CC, CC, 1);

    // attention -> abf (bf16 [M, 768])
    attn_kernel<<<dim3(33, HH, BB), 32, 0, stream>>>(big, abf);

    // proj + bias + residual -> xf (fp32)
    gemm_bf16_kernel<<<dim3(MM / 32, CC / 256), 256, 0, stream>>>(
        abf, CC, wpb, CC, b_proj, xf, xf, nullptr, CC, CC, 0);

    ln_kernel<<<MM, 256, 0, stream>>>(xf, g2, b2, abf);

    // fc1 + bias + exact GELU -> bf16 [M, 3072]
    gemm_bf16_kernel<<<dim3(MM / 32, HID / 256), 256, 0, stream>>>(
        abf, CC, w1b, CC, b_fc1, nullptr, nullptr, big, HID, CC, 2);

    // fc2 + bias + residual -> d_out (fp32)
    gemm_bf16_kernel<<<dim3(MM / 32, CC / 256), 256, 0, stream>>>(
        big, HID, w2b, HID, b_fc2, xf, (float*)d_out, nullptr, CC, HID, 0);
}
